// Model_14998025797929
// MI455X (gfx1250) — matrix operand order repaired and checked by reading, not compiled, not run on hardware
//
#include <hip/hip_runtime.h>
#include <hip/hip_bf16.h>
#include <math.h>

typedef __bf16 bf16;
typedef __attribute__((ext_vector_type(16))) __bf16 v16bf;
typedef __attribute__((ext_vector_type(8)))  __bf16 v8bf;
typedef __attribute__((ext_vector_type(8)))  float  v8f;

#define B_   64
#define T_   512
#define DIN_ 256
#define N_   1024
#define O_   128

#define STEP_BLOCKS 32   // 32 blocks x 8 waves = 256 waves = all 16x16 tiles/step

__device__ __forceinline__ float fast_tanh(float x) {
#if __has_builtin(__builtin_amdgcn_tanhf)
  return __builtin_amdgcn_tanhf(x);   // gfx1250 V_TANH_F32 transcendental
#else
  return tanhf(x);
#endif
}

// ---------------------------------------------------------------------------
// Pack a row-major f32 [K][Ncols] matrix into the WMMA B-operand tile layout
// for v_wmma_f32_16x16x32_bf16 (wave32):
//   tile (nt, kb) covers K rows kb*32..kb*32+31, N cols nt*16..nt*16+15
//   lane l holds column n = nt*16 + (l&15), K-half (l>>4): 16 contiguous bf16
//   flat index = ((nt*KB + kb)*32 + lane)*16 + j, element (kb*32+(l>>4)*16+j, n)
// ---------------------------------------------------------------------------
__global__ void pack_b_kernel(const float* __restrict__ src, bf16* __restrict__ dst,
                              int K, int Ncols) {
  int tid = blockIdx.x * blockDim.x + threadIdx.x;
  if (tid >= K * Ncols) return;
  int KB   = K >> 5;
  int j    = tid & 15;
  int lane = (tid >> 4) & 31;
  int kb   = (tid >> 9) % KB;
  int nt   = (tid >> 9) / KB;
  int k    = kb * 32 + ((lane >> 4) << 4) + j;
  int n    = nt * 16 + (lane & 15);
  dst[tid] = (bf16)src[(size_t)k * Ncols + n];
}

__global__ void cvt_bf16_kernel(const float* __restrict__ src, bf16* __restrict__ dst, int n) {
  int i = blockIdx.x * blockDim.x + threadIdx.x;
  if (i < n) dst[i] = (bf16)src[i];
}

// s[:,0,:] = 0 in d_out, zero the bf16 ping-pong state buffer, reset barrier.
__global__ void init_state_kernel(bf16* __restrict__ s0, float* __restrict__ s_out,
                                  unsigned* __restrict__ bar) {
  int i = blockIdx.x * blockDim.x + threadIdx.x;   // i = m*1024 + n
  if (i == 0) { bar[0] = 0u; bar[1] = 0u; }        // counter, flag
  if (i >= B_ * N_) return;
  int m = i >> 10, n = i & 1023;
  s0[i] = (bf16)0.f;
  s_out[((size_t)m * T_ + 0) * N_ + n] = 0.f;
}

// A-operand loader: 16-bit A 16x32 layout. Lane (hi = l>>4, lo = l&15) holds
// row M = lo, K = hi*16 + 0..15 (contiguous, matching packed B).
// Caller passes p already offset by (row*ld + kchunk*32 + hi*16).
__device__ __forceinline__ v16bf load_a_bf16(const bf16* p) {
  v8bf a0 = *(const v8bf*)(p);
  v8bf a1 = *(const v8bf*)(p + 8);
  v16bf a;
#pragma unroll
  for (int i = 0; i < 8; ++i) { a[i] = a0[i]; a[i + 8] = a1[i]; }
  return a;
}
__device__ __forceinline__ v16bf load_a_f32(const float* p) {
  v16bf a;
#pragma unroll
  for (int i = 0; i < 8; ++i) { a[i] = (bf16)p[i]; a[i + 8] = (bf16)p[i + 8]; }
  return a;
}

// Grid-wide release/acquire barrier (32 co-resident blocks, generation = t).
__device__ __forceinline__ void grid_barrier(unsigned* __restrict__ counter,
                                             volatile unsigned* __restrict__ flag,
                                             unsigned gen) {
  __syncthreads();
  if (threadIdx.x == 0) {
    __threadfence();                                  // release my state writes
    unsigned old = atomicAdd(counter, 1u);
    if (old == (unsigned)STEP_BLOCKS - 1u) {
      atomicExch(counter, 0u);                        // reset before release
      __threadfence();
      *flag = gen;
    } else {
      while (*flag < gen) { __builtin_amdgcn_s_sleep(2); }
    }
  }
  __syncthreads();
  __threadfence();                                    // acquire others' writes
}

// ---------------------------------------------------------------------------
// Persistent recurrence kernel: ONE launch for all 511 steps.
//   s[t] = tanh(s[t-1] @ W + X[:, t-1] @ Win)
// 256 waves, one fixed 16x16 tile per wave (4 M-tiles x 64 N-tiles).
// Per step: 32 WMMAs over W (K=1024) + 8 WMMAs over Win (K=256), then one
// device-wide barrier. All operands (W 2MB, Win 0.5MB, state 128KB) stay
// L2-resident; this removes ~510 kernel-launch latencies from the serial path.
// ---------------------------------------------------------------------------
__global__ void __launch_bounds__(256)
rnn_persistent_kernel(const bf16* __restrict__ Wp, const bf16* __restrict__ Winp,
                      const bf16* __restrict__ Xb, bf16* __restrict__ sb0,
                      bf16* __restrict__ sb1, float* __restrict__ s_out,
                      unsigned* __restrict__ bar) {
  int wave = (blockIdx.x * blockDim.x + threadIdx.x) >> 5;
  int lane = threadIdx.x & 31;
  int mT = wave & 3;            // 64 rows  -> 4 M-tiles
  int nT = wave >> 2;           // 1024 cols -> 64 N-tiles
  int hi = lane >> 4, lo = lane & 15;

  unsigned* counter = bar;
  volatile unsigned* flag = bar + 1;

  const bf16* bp  = Wp   + ((size_t)nT * 32 * 32 + lane) * 16;  // KB(W) = 32
  const bf16* bp2 = Winp + ((size_t)nT * 8  * 32 + lane) * 16;  // KB(Win) = 8
  const int n = nT * 16 + lo;

  for (int t = 1; t < T_; ++t) {
    const bf16* sp = (t & 1) ? sb0 : sb1;
    bf16*       sn = (t & 1) ? sb1 : sb0;

    v8f acc = {0.f, 0.f, 0.f, 0.f, 0.f, 0.f, 0.f, 0.f};

    // ---- s[t-1] @ W, K = 1024 (32 chunks of 32) ----
    const bf16* arow = sp + (size_t)(mT * 16 + lo) * N_ + hi * 16;
#pragma unroll 4
    for (int kb = 0; kb < 32; ++kb) {
      v16bf a = load_a_bf16(arow + kb * 32);
      v16bf b = *(const v16bf*)(bp + (size_t)kb * 32 * 16);
      acc = __builtin_amdgcn_wmma_f32_16x16x32_bf16(false, a, false, b,
                                                    (short)0, acc, false, false);
    }

    // ---- X[:, t-1] @ Win, K = 256 (8 chunks of 32) ----
    const bf16* xrow = Xb + ((size_t)(mT * 16 + lo) * T_ + (t - 1)) * DIN_ + hi * 16;
#pragma unroll
    for (int kb = 0; kb < 8; ++kb) {
      v16bf a = load_a_bf16(xrow + kb * 32);
      v16bf b = *(const v16bf*)(bp2 + (size_t)kb * 32 * 16);
      acc = __builtin_amdgcn_wmma_f32_16x16x32_bf16(false, a, false, b,
                                                    (short)0, acc, false, false);
    }

    // ---- epilogue: tanh + store (C/D layout: VGPR r -> row r + hi*8) ----
#pragma unroll
    for (int r = 0; r < 8; ++r) {
      int m = mT * 16 + r + hi * 8;
      float v = fast_tanh(acc[r]);
      s_out[((size_t)m * T_ + t) * N_ + n] = v;
      sn[(size_t)m * N_ + n] = (bf16)v;
    }

    grid_barrier(counter, flag, (unsigned)t);
  }
}

// ---------------------------------------------------------------------------
// out = s @ Wout + bout over all B*T rows. The whole packed Wout (256 KB,
// identical for every wave) is staged once per block into LDS (CDNA5 allows
// up to 320 KB/workgroup), cutting B-operand L2 traffic 8x; inner loop then
// mixes ds_load (B) with global f32 loads of s (read exactly once, converted
// to bf16 in registers). One wave per 16-row stripe, 8 accumulators = 16x128.
// ---------------------------------------------------------------------------
__global__ void __launch_bounds__(256)
out_proj_kernel(const float* __restrict__ s, const bf16* __restrict__ Woutp,
                const float* __restrict__ bout, float* __restrict__ out) {
  extern __shared__ char smem[];
  bf16* bs = (bf16*)smem;                             // 131072 bf16 = 256 KB

  // cooperative preload: 256 threads x 64 x 16B = 256 KB
  {
    const uint4* srcv = (const uint4*)Woutp;
    uint4* dstv = (uint4*)bs;
#pragma unroll 4
    for (int i = threadIdx.x; i < (N_ * O_) / 8; i += 256) dstv[i] = srcv[i];
  }
  __syncthreads();

  int wave = (blockIdx.x * blockDim.x + threadIdx.x) >> 5;  // 0..2047 M-tiles
  int lane = threadIdx.x & 31;
  int hi = lane >> 4, lo = lane & 15;

  v8f acc[8];
#pragma unroll
  for (int nt = 0; nt < 8; ++nt) acc[nt] = (v8f){0.f,0.f,0.f,0.f,0.f,0.f,0.f,0.f};

  const float* arow = s + (size_t)(wave * 16 + lo) * N_ + hi * 16;
  for (int kb = 0; kb < 32; ++kb) {
    v16bf a = load_a_f32(arow + kb * 32);
#pragma unroll
    for (int nt = 0; nt < 8; ++nt) {
      v16bf b = *(const v16bf*)(bs + (((size_t)nt * 32 + kb) * 32 + lane) * 16);
      acc[nt] = __builtin_amdgcn_wmma_f32_16x16x32_bf16(false, a, false, b,
                                                        (short)0, acc[nt], false, false);
    }
  }
#pragma unroll
  for (int nt = 0; nt < 8; ++nt) {
    float bv = bout[nt * 16 + lo];
#pragma unroll
    for (int r = 0; r < 8; ++r) {
      int row = wave * 16 + r + hi * 8;                  // flat b*T + t
      out[(size_t)row * O_ + nt * 16 + lo] = acc[nt][r] + bv;
    }
  }
}

extern "C" void kernel_launch(void* const* d_in, const int* in_sizes, int n_in,
                              void* d_out, int out_size, void* d_ws, size_t ws_size,
                              hipStream_t stream) {
  const float* X    = (const float*)d_in[0];   // [B,T,DIN]
  const float* Win  = (const float*)d_in[1];   // [DIN,N]
  const float* W    = (const float*)d_in[2];   // [N,N]
  const float* Wout = (const float*)d_in[3];   // [N,O]
  const float* bout = (const float*)d_in[4];   // [O]

  float* s_out = (float*)d_out;                        // [B,T,N]
  float* out   = (float*)d_out + (size_t)B_ * T_ * N_; // [B,T,O]

  // workspace layout (~20.6 MB total, all 256B-aligned)
  char* ws = (char*)d_ws;
  bf16* Wp    = (bf16*)(ws);                                     // 2   MB
  bf16* Winp  = (bf16*)(ws + (size_t)(2u << 20));                // 0.5 MB
  bf16* Woutp = (bf16*)(ws + (size_t)(2u << 20) + (512u << 10)); // 0.25 MB
  bf16* Xb    = (bf16*)(ws + (size_t)(3u << 20));                // 16  MB
  bf16* sb0   = (bf16*)(ws + (size_t)(20u << 20));               // 128 KB
  bf16* sb1   = (bf16*)(ws + (size_t)(20u << 20) + (128u << 10));// 128 KB
  unsigned* bar = (unsigned*)(ws + (size_t)(20u << 20) + (512u << 10)); // 8 B

  // one-time packing/conversion (weights stay L2-resident afterwards)
  pack_b_kernel<<<(N_ * N_)   / 256, 256, 0, stream>>>(W,    Wp,    N_,   N_);
  pack_b_kernel<<<(DIN_ * N_) / 256, 256, 0, stream>>>(Win,  Winp,  DIN_, N_);
  pack_b_kernel<<<(N_ * O_)   / 256, 256, 0, stream>>>(Wout, Woutp, N_,   O_);
  cvt_bf16_kernel<<<(B_ * T_ * DIN_) / 256, 256, 0, stream>>>(X, Xb, B_ * T_ * DIN_);
  init_state_kernel<<<(B_ * N_) / 256, 256, 0, stream>>>(sb0, s_out, bar);

  // entire 511-step recurrence in ONE persistent launch (grid barrier/step)
  rnn_persistent_kernel<<<STEP_BLOCKS, 256, 0, stream>>>(Wp, Winp, Xb, sb0, sb1,
                                                         s_out, bar);

  // fully parallel output projection: 256 blocks, 256 KB dynamic LDS each
  out_proj_kernel<<<(B_ * T_ / 16) * 32 / 256, 256, (N_ * O_) * 2, stream>>>(
      s_out, Woutp, bout, out);
}
